// RN_66348654789238
// MI455X (gfx1250) — compile-verified, matrix operand order checked by reading, not run on hardware
//
#include <hip/hip_runtime.h>

#define MB    2048
#define DNUM  25
#define NPAIR 625        // D*D
#define H     256
#define NCLS  10
#define TPB   64         // 2 waves per workgroup
#define PTILES_PER_WAVE 10   // each wave: 10 x 32-row pair-tiles (2 waves * 10 * 32 = 640)
#define ROWSTRIDE 264    // bf16 elements per tile row (256 + 8 pad, 16B aligned rows)

typedef __attribute__((ext_vector_type(16))) __bf16 v16bf;
typedef __attribute__((ext_vector_type(8)))  float  v8f;

__device__ __forceinline__ unsigned short f2bf(float x) {
    unsigned int u = __float_as_uint(x);
    unsigned int r = u + 0x7FFFu + ((u >> 16) & 1u);   // round-to-nearest-even
    return (unsigned short)(r >> 16);
}

union Frag { uint4 u[2]; v16bf v; };

// ---------------------------------------------------------------------------
// Pack g2/g3/g4 weights (f32, [256][256] row-major, fan_in x fan_out) into
// bf16 WMMA B-operand fragments.  Fragment (layer, nt, kt) holds the 32x16
// (KxN) tile; per lane 16 bf16 = 32 bytes, laid out to match the CDNA5
// 16-bit operand layout: lane&15 = column, lanes>=16 take the +8 K half,
// VGPR v covers K pairs {2v} (v<4) / {16+2(v-4)} (v>=4).
// ---------------------------------------------------------------------------
__global__ void __launch_bounds__(256)
rn_pack_kernel(const float* __restrict__ g2w, const float* __restrict__ g3w,
               const float* __restrict__ g4w, unsigned short* __restrict__ wpack)
{
    int idx = blockIdx.x * blockDim.x + threadIdx.x;
    const int total = 3 * 16 * 8 * 32 * 16;           // 196608 elements
    if (idx >= total) return;
    int e     =  idx        & 15;
    int lane  = (idx >> 4)  & 31;
    int kt    = (idx >> 9)  & 7;
    int nt    = (idx >> 12) & 15;
    int layer =  idx >> 16;
    int v = e >> 1, h = e & 1;
    int klocal = ((v < 4) ? (2 * v) : (16 + 2 * (v - 4))) + h + ((lane >= 16) ? 8 : 0);
    int k = kt * 32 + klocal;
    int n = nt * 16 + (lane & 15);
    const float* w = (layer == 0) ? g2w : (layer == 1) ? g3w : g4w;
    wpack[idx] = f2bf(w[k * H + n]);
}

// ---------------------------------------------------------------------------
// One workgroup per sample: fused pair-construction + g-MLP (WMMA bf16) +
// pair-sum + f-MLP + log_softmax.  Each wave works on 32-row pair-tiles so
// every B fragment load feeds two WMMAs (halves L2 weight traffic).
// ---------------------------------------------------------------------------
__global__ void __launch_bounds__(TPB, 3)
rn_main_kernel(const float* __restrict__ x,    const float* __restrict__ coord,
               const float* __restrict__ g1w,  const float* __restrict__ g1b,
               const float* __restrict__ g2b,  const float* __restrict__ g3b,
               const float* __restrict__ g4b,
               const unsigned short* __restrict__ wpack,
               const float* __restrict__ f1w,  const float* __restrict__ f1b,
               const float* __restrict__ fc2w, const float* __restrict__ fc2b,
               const float* __restrict__ fc3w, const float* __restrict__ fc3b,
               float* __restrict__ out)
{
    __shared__ float s_xf[DNUM * 3];                       // value + 2 coords
    __shared__ float s_w1[6 * H];
    __shared__ float s_b1[H];
    __shared__ float s_b[3][H];
    __shared__ __align__(16) unsigned short s_tile[2][32 * ROWSTRIDE];
    __shared__ float s_colsum[H];
    __shared__ float s_f[2][H];
    __shared__ float s_logits[16];

    const int tid  = threadIdx.x;
    const int wave = tid >> 5;
    const int lane = tid & 31;
    const int mb   = blockIdx.x;

    // ---- workgroup-wide preload (tiny) ----
    for (int i = tid; i < 6 * H; i += TPB) s_w1[i] = g1w[i];
    for (int i = tid; i < H; i += TPB) {
        s_b1[i]   = g1b[i];
        s_b[0][i] = g2b[i];
        s_b[1][i] = g3b[i];
        s_b[2][i] = g4b[i];
        s_colsum[i] = 0.0f;
    }
    for (int i = tid; i < DNUM; i += TPB) {
        s_xf[i * 3 + 0] = x[mb * DNUM + i];
        s_xf[i * 3 + 1] = coord[(mb * DNUM + i) * 2 + 0];
        s_xf[i * 3 + 2] = coord[(mb * DNUM + i) * 2 + 1];
    }
    __syncthreads();

    unsigned short* tile = s_tile[wave];
    const int klo  = (lane >= 16) ? 8 : 0;   // K-half offset (also D-row half)
    const int nloc = lane & 15;
    // Hoisted per-lane invariant addresses (sub-tile 0 at rows 0-15, sub-tile 1 at 16-31).
    unsigned short* const a_base0 = &tile[(lane & 15) * ROWSTRIDE + klo];
    unsigned short* const a_base1 = a_base0 + 16 * ROWSTRIDE;
    unsigned short* const d_base0 = &tile[klo * ROWSTRIDE + nloc];
    unsigned short* const d_base1 = d_base0 + 16 * ROWSTRIDE;
    const uint4* const wlane = (const uint4*)wpack + lane * 2;

#pragma unroll 1
    for (int ti = 0; ti < PTILES_PER_WAVE; ++ti) {
        const int tp   = wave * PTILES_PER_WAVE + ti;   // 0..19
        const int row0 = tp * 32;

        // ---- g1 (K=6) on VALU, straight into the LDS tile as bf16 ----
#pragma unroll 1
        for (int r = 0; r < 32; ++r) {
            int p = row0 + r;
            float f0 = 0.f, f1 = 0.f, f2 = 0.f, f3 = 0.f, f4 = 0.f, f5 = 0.f;
            if (p < NPAIR) {
                int a = p / DNUM, b = p - a * DNUM;
                f0 = s_xf[b * 3]; f1 = s_xf[b * 3 + 1]; f2 = s_xf[b * 3 + 2];
                f3 = s_xf[a * 3]; f4 = s_xf[a * 3 + 1]; f5 = s_xf[a * 3 + 2];
            }
#pragma unroll
            for (int cc = 0; cc < H / 32; ++cc) {
                int c = cc * 32 + lane;
                float acc = s_b1[c]
                          + f0 * s_w1[0 * H + c] + f1 * s_w1[1 * H + c]
                          + f2 * s_w1[2 * H + c] + f3 * s_w1[3 * H + c]
                          + f4 * s_w1[4 * H + c] + f5 * s_w1[5 * H + c];
                acc = fmaxf(acc, 0.0f);
                if (p >= NPAIR) acc = 0.0f;          // padded rows
                tile[r * ROWSTRIDE + c] = f2bf(acc);
            }
        }

        // ---- g2, g3, g4 : 32x256 pair-tile @ 256x256 via bf16 WMMA ----
#pragma unroll 1
        for (int layer = 0; layer < 3; ++layer) {
            // Load full A operand (32 rows x 256 K) into registers first.
            Frag A0[8], A1[8];
#pragma unroll
            for (int kt = 0; kt < 8; ++kt) {
                const unsigned short* ap0 = a_base0 + kt * 32;
                const unsigned short* ap1 = a_base1 + kt * 32;
                A0[kt].u[0] = *(const uint4*)(ap0);
                A0[kt].u[1] = *(const uint4*)(ap0 + 16);
                A1[kt].u[0] = *(const uint4*)(ap1);
                A1[kt].u[1] = *(const uint4*)(ap1 + 16);
            }
            const bool last = (layer == 2);
            const uint4* wfrag0 = wlane + (size_t)(layer * 16) * 8 * 64;
#pragma unroll 1
            for (int nt = 0; nt < 16; ++nt) {
                // Prefetch bias, then issue the whole B-fragment clause so the
                // WMMA chain only needs incremental loadcnt waits.
                float bias = s_b[layer][nt * 16 + nloc];
                const uint4* wfrag = wfrag0 + (size_t)nt * 8 * 64;
                Frag B[8];
#pragma unroll
                for (int kt = 0; kt < 8; ++kt) {
                    B[kt].u[0] = wfrag[kt * 64 + 0];
                    B[kt].u[1] = wfrag[kt * 64 + 1];
                }
                // Keep all 16 loads issued ahead of the WMMA chain.
                __builtin_amdgcn_sched_barrier(0);
                v8f acc0 = { bias, bias, bias, bias, bias, bias, bias, bias };
                v8f acc1 = acc0;
#pragma unroll
                for (int kt = 0; kt < 8; ++kt) {
                    acc0 = __builtin_amdgcn_wmma_f32_16x16x32_bf16(
                        false, A0[kt].v, false, B[kt].v, (short)0, acc0, false, false);
                    acc1 = __builtin_amdgcn_wmma_f32_16x16x32_bf16(
                        false, A1[kt].v, false, B[kt].v, (short)0, acc1, false, false);
                }
                if (!last) {
                    // ReLU -> bf16 -> LDS (D layout: m = v + klo, n = nt*16 + nloc)
#pragma unroll
                    for (int v = 0; v < 8; ++v) {
                        float r0 = fmaxf(acc0[v], 0.0f);
                        float r1 = fmaxf(acc1[v], 0.0f);
                        d_base0[v * ROWSTRIDE + nt * 16] = f2bf(r0);
                        d_base1[v * ROWSTRIDE + nt * 16] = f2bf(r1);
                    }
                } else {
                    // g4: ReLU then masked sum over the pair-tile's valid rows
                    float s = 0.0f;
#pragma unroll
                    for (int v = 0; v < 8; ++v) {
                        int m = v + klo;
                        float r0 = fmaxf(acc0[v], 0.0f);
                        float r1 = fmaxf(acc1[v], 0.0f);
                        s += (row0 + m      < NPAIR) ? r0 : 0.0f;
                        s += (row0 + 16 + m < NPAIR) ? r1 : 0.0f;
                    }
                    atomicAdd(&s_colsum[nt * 16 + nloc], s);
                }
            }
        }
    }
    __syncthreads();

    // ---- f-MLP (tiny, f32 on VALU; weight reads coalesced across threads) ----
    for (int c = tid; c < H; c += TPB) {
        float acc = f1b[c];
#pragma unroll 4
        for (int k = 0; k < H; ++k) acc += s_colsum[k] * f1w[k * H + c];
        s_f[0][c] = fmaxf(acc, 0.0f);
    }
    __syncthreads();
    for (int c = tid; c < H; c += TPB) {
        float acc = fc2b[c];
#pragma unroll 4
        for (int k = 0; k < H; ++k) acc += s_f[0][k] * fc2w[k * H + c];
        s_f[1][c] = fmaxf(acc, 0.0f);
    }
    __syncthreads();
    if (tid < NCLS) {
        float acc = fc3b[tid];
#pragma unroll 4
        for (int k = 0; k < H; ++k) acc += s_f[1][k] * fc3w[k * NCLS + tid];
        s_logits[tid] = acc;
    }
    __syncthreads();
    if (tid == 0) {
        float mx = s_logits[0];
        for (int i = 1; i < NCLS; ++i) mx = fmaxf(mx, s_logits[i]);
        float se = 0.0f;
        for (int i = 0; i < NCLS; ++i) se += expf(s_logits[i] - mx);
        float lse = mx + logf(se);
        for (int i = 0; i < NCLS; ++i) out[mb * NCLS + i] = s_logits[i] - lse;
    }
}

extern "C" void kernel_launch(void* const* d_in, const int* in_sizes, int n_in,
                              void* d_out, int out_size, void* d_ws, size_t ws_size,
                              hipStream_t stream) {
    const float* x    = (const float*)d_in[0];
    const float* co   = (const float*)d_in[1];
    const float* g1w  = (const float*)d_in[2];
    const float* g1b  = (const float*)d_in[3];
    const float* g2w  = (const float*)d_in[4];
    const float* g2b  = (const float*)d_in[5];
    const float* g3w  = (const float*)d_in[6];
    const float* g3b  = (const float*)d_in[7];
    const float* g4w  = (const float*)d_in[8];
    const float* g4b  = (const float*)d_in[9];
    const float* f1w  = (const float*)d_in[10];
    const float* f1b  = (const float*)d_in[11];
    const float* fc2w = (const float*)d_in[12];
    const float* fc2b = (const float*)d_in[13];
    const float* fc3w = (const float*)d_in[14];
    const float* fc3b = (const float*)d_in[15];
    float* out = (float*)d_out;
    unsigned short* wpack = (unsigned short*)d_ws;   // 384 KB used

    hipLaunchKernelGGL(rn_pack_kernel, dim3(768), dim3(256), 0, stream,
                       g2w, g3w, g4w, wpack);
    hipLaunchKernelGGL(rn_main_kernel, dim3(MB), dim3(TPB), 0, stream,
                       x, co, g1w, g1b, g2b, g3b, g4b, wpack,
                       f1w, f1b, fc2w, fc2b, fc3w, fc3b, out);
}
